// CEAlignmentInformation_66374424592633
// MI455X (gfx1250) — matrix operand order repaired and checked by reading, not compiled, not run on hardware
//
#include <hip/hip_runtime.h>
#include <hip/hip_bf16.h>
#include <math.h>

// ---------------- problem constants (from reference) ----------------
#define B_N   2048
#define D1_N  512
#define D2_N  512
#define H_N   1024
#define E_N   64
#define L_N   10
#define Q_N   (L_N * E_N)   // 640
#define EPSF  1e-8f
#define ATOLF 0.01f
#define MAX_ITERS 500

typedef __attribute__((ext_vector_type(8)))  _Float16 v8h;
typedef __attribute__((ext_vector_type(16))) _Float16 v16h;
typedef __attribute__((ext_vector_type(8)))  float    v8f;

union V16U { v16h v; v8h h[2]; };

// CDNA5 async global->LDS path (ASYNCcnt), with graceful fallback.
#if defined(__HIP_DEVICE_COMPILE__) && defined(__gfx1250__) && \
    __has_builtin(__builtin_amdgcn_global_load_async_to_lds_b128) && \
    __has_builtin(__builtin_amdgcn_s_wait_asynccnt)
#define USE_ASYNC_LDS 1
#else
#define USE_ASYNC_LDS 0
#endif

#define AS1 __attribute__((address_space(1)))
#define AS3 __attribute__((address_space(3)))
typedef int async_v4i __attribute__((vector_size(16)));   // builtin's b128 payload type

// ---------------- f32 -> f16 conversion ----------------
__global__ void f32_to_f16_kernel(const float* __restrict__ in, _Float16* __restrict__ out, int n) {
    int i = blockIdx.x * 256 + threadIdx.x;
    if (i < n) out[i] = (_Float16)in[i];
}

// ---------------- WMMA GEMM ----------------
// Computes, per batch z:  C[M x N] = post( A[M x K] * B[N x K]^T )
// A rows contiguous in K (stride lda halves), B rows contiguous in K (stride ldb) == W row-major.
// mode: 0 = acc + bias ; 1 = relu(acc + bias) ; 2 = exp(scale * acc)
#define BM 128
#define BN 128
#define BK 32
#define LDSS 40   // padded LDS row stride in halves (80B -> conflict-free b128 reads)

__global__ __launch_bounds__(256)
void wmma_gemm_kernel(const _Float16* __restrict__ A, int lda, long long strideAb,
                      const _Float16* __restrict__ Bm, int ldb, long long strideBb,
                      const float* __restrict__ bias,
                      float* __restrict__ outF, _Float16* __restrict__ outH,
                      int ldc, long long strideCb,
                      int K, int mode, float scale)
{
    // double-buffered tiles: 2 x (128 x 40 halves) x 2 arrays = 40KB of 320KB/WGP
    __shared__ __align__(16) _Float16 lA[2][BM * LDSS];
    __shared__ __align__(16) _Float16 lB[2][BN * LDSS];

    const int bz = blockIdx.z;
    A  += (long long)bz * strideAb;
    Bm += (long long)bz * strideBb;
    const long long cbase = (long long)bz * strideCb;

    const int t    = threadIdx.x;
    const int wave = t >> 5;
    const int lane = t & 31;
    const int wm   = wave >> 2;   // 0..1  (64 rows each)
    const int wn   = wave & 3;    // 0..3  (32 cols each)
    const int m0   = blockIdx.y * BM;
    const int n0   = blockIdx.x * BN;

    v8f zero = {};
    v8f acc[4][2];
    for (int i = 0; i < 4; ++i)
        for (int j = 0; j < 2; ++j)
            acc[i][j] = zero;

    // per-thread fixed staging slots: row0 in [0,63], row0+64, seg in [0,3]
    const int row0 = t >> 2;
    const int seg  = t & 3;
    const _Float16* gA0 = A  + (long long)(m0 + row0)      * lda + seg * 8;
    const _Float16* gA1 = A  + (long long)(m0 + row0 + 64) * lda + seg * 8;
    const _Float16* gB0 = Bm + (long long)(n0 + row0)      * ldb + seg * 8;
    const _Float16* gB1 = Bm + (long long)(n0 + row0 + 64) * ldb + seg * 8;
    const int sA0 = row0 * LDSS + seg * 8;
    const int sA1 = (row0 + 64) * LDSS + seg * 8;

    auto stage = [&](int buf, int kt) {
#if USE_ASYNC_LDS
        __builtin_amdgcn_global_load_async_to_lds_b128(
            (AS1 async_v4i*)(gA0 + kt), (AS3 async_v4i*)&lA[buf][sA0], 0, 0);
        __builtin_amdgcn_global_load_async_to_lds_b128(
            (AS1 async_v4i*)(gA1 + kt), (AS3 async_v4i*)&lA[buf][sA1], 0, 0);
        __builtin_amdgcn_global_load_async_to_lds_b128(
            (AS1 async_v4i*)(gB0 + kt), (AS3 async_v4i*)&lB[buf][sA0], 0, 0);
        __builtin_amdgcn_global_load_async_to_lds_b128(
            (AS1 async_v4i*)(gB1 + kt), (AS3 async_v4i*)&lB[buf][sA1], 0, 0);
#else
        *(v8h*)&lA[buf][sA0] = *(const v8h*)(gA0 + kt);
        *(v8h*)&lA[buf][sA1] = *(const v8h*)(gA1 + kt);
        *(v8h*)&lB[buf][sA0] = *(const v8h*)(gB0 + kt);
        *(v8h*)&lB[buf][sA1] = *(const v8h*)(gB1 + kt);
#endif
    };

    const int lrow = lane & 15;
    const int kA0  = (lane >> 4) * 8;    // A: K in {k0..k0+7} U {k0+16..k0+23}
    const int kB0  = (lane >> 4) * 8;    // B: K in {k0..k0+7} U {k0+16..k0+23}

    const int nt = K / BK;
    stage(0, 0);
    for (int it = 0; it < nt; ++it) {
        const int b = it & 1;
        const bool hasNext = (it + 1 < nt);
        if (hasNext) stage(b ^ 1, (it + 1) * BK);   // overlaps compute of tile `it`
#if USE_ASYNC_LDS
        // async loads complete in order: <=4 outstanding leaves only the
        // just-issued next-tile loads in flight, current tile is resident.
        if (hasNext) __builtin_amdgcn_s_wait_asynccnt(4);
        else         __builtin_amdgcn_s_wait_asynccnt(0);
#endif
        __syncthreads();

        const _Float16* tA = lA[b];
        const _Float16* tB = lB[b];
        for (int i = 0; i < 4; ++i) {
            int ra = wm * 64 + i * 16 + lrow;
            V16U fa;
            fa.h[0] = *(const v8h*)&tA[ra * LDSS + kA0];
            fa.h[1] = *(const v8h*)&tA[ra * LDSS + kA0 + 16];
            for (int j = 0; j < 2; ++j) {
                int rb = wn * 32 + j * 16 + lrow;
                V16U fb;
                fb.h[0] = *(const v8h*)&tB[rb * LDSS + kB0];
                fb.h[1] = *(const v8h*)&tB[rb * LDSS + kB0 + 16];
                acc[i][j] = __builtin_amdgcn_wmma_f32_16x16x32_f16(
                    false, fa.v, false, fb.v, (short)0, acc[i][j], false, false);
            }
        }
        __syncthreads();   // all waves done reading buf b before it is restaged
    }

    // C/D layout: VGPR e -> M = e (lanes 0-15) / e+8 (lanes 16-31); N = lane&15
    const int colOff  = lane & 15;
    const int rowHalf = (lane >> 4) * 8;
    for (int i = 0; i < 4; ++i) {
        for (int j = 0; j < 2; ++j) {
            int col = n0 + wn * 32 + j * 16 + colOff;
            float bv = (bias != nullptr) ? bias[col] : 0.f;
            for (int e = 0; e < 8; ++e) {
                int row = m0 + wm * 64 + i * 16 + e + rowHalf;
                float a_ = acc[i][j][e];
                float v;
                if (mode == 2) v = __expf(a_ * scale);
                else { v = a_ + bv; if (mode == 1) v = fmaxf(v, 0.f); }
                long long idx = cbase + (long long)row * ldc + col;
                if (outF) outF[idx] = v;
                if (outH) outH[idx] = (_Float16)v;
            }
        }
    }
}

// ---------------- discriminator: softmax(x @ W^T + b), L=10 ----------------
__global__ __launch_bounds__(256)
void disc_softmax_kernel(const float* __restrict__ x1, int K1,
                         const float* __restrict__ x2, int K2,
                         const float* __restrict__ W, const float* __restrict__ b,
                         float* __restrict__ out)
{
    int a = blockIdx.x, t = threadIdx.x;
    int K = K1 + K2;
    float acc[L_N];
    for (int l = 0; l < L_N; ++l) acc[l] = 0.f;
    for (int k = t; k < K; k += 256) {
        float xv = (k < K1) ? x1[(long long)a * K1 + k] : x2[(long long)a * K2 + (k - K1)];
        for (int l = 0; l < L_N; ++l) acc[l] += xv * W[(long long)l * K + k];
    }
    __shared__ float red[256];
    __shared__ float logits[L_N];
    for (int l = 0; l < L_N; ++l) {
        red[t] = acc[l]; __syncthreads();
        for (int s = 128; s > 0; s >>= 1) { if (t < s) red[t] += red[t + s]; __syncthreads(); }
        if (t == 0) logits[l] = red[0] + b[l];
        __syncthreads();
    }
    if (t == 0) {
        float mx = logits[0];
        for (int l = 1; l < L_N; ++l) mx = fmaxf(mx, logits[l]);
        float s = 0.f;
        for (int l = 0; l < L_N; ++l) { float e = __expf(logits[l] - mx); logits[l] = e; s += e; }
        for (int l = 0; l < L_N; ++l) out[(long long)a * L_N + l] = logits[l] / s;
    }
}

// ---------------- per-(a,l) row normalize over E=64 (ddof=1) ----------------
__global__ __launch_bounds__(64)
void norm_rows_kernel(const float* __restrict__ q, _Float16* __restrict__ outH)
{
    int row = blockIdx.x, t = threadIdx.x;
    float v = q[(long long)row * E_N + t];
    __shared__ float s1[64], s2[64];
    s1[t] = v; s2[t] = v * v; __syncthreads();
    for (int s = 32; s > 0; s >>= 1) {
        if (t < s) { s1[t] += s1[t + s]; s2[t] += s2[t + s]; }
        __syncthreads();
    }
    float mean = s1[0] / (float)E_N;
    float var  = (s2[0] - (float)E_N * mean * mean) / (float)(E_N - 1);
    outH[(long long)row * E_N + t] = (_Float16)((v - mean) * rsqrtf(var + EPSF));
}

// ---------------- Sinkhorn ----------------
// m stored [L][B][B] f32; 168MB -> resident in 192MB L2 across the iteration loop.
__global__ void sink_init_colsum_kernel(const float* __restrict__ m, float* __restrict__ c,
                                        int* __restrict__ done)
{
    int l = blockIdx.x, chunk = blockIdx.y, t = threadIdx.x;
    int b = chunk * 256 + t;
    const float* ml = m + (long long)l * B_N * B_N;
    float s = 0.f;
    for (int a = 0; a < B_N; ++a) s += ml[(long long)a * B_N + b];
    c[l * B_N + b] = s;
    if (chunk == 0 && t == 0) done[l] = 0;
}

__global__ void sink_reset_kernel(float* __restrict__ c2, int* __restrict__ viol1)
{
    int l = blockIdx.x, t = threadIdx.x;
    for (int b = t; b < B_N; b += 256) c2[l * B_N + b] = 0.f;
    if (t == 0) viol1[l] = 0;
}

// r[l][a] = sum_b m * s2[b]; stop1 violation check
__global__ __launch_bounds__(256)
void sink_rowsum_kernel(const float* __restrict__ m, const float* __restrict__ c,
                        const float* __restrict__ py1, const float* __restrict__ py2,
                        float* __restrict__ r, int* __restrict__ viol1,
                        const int* __restrict__ done)
{
    int l = blockIdx.x;
    if (done[l]) return;
    __shared__ float s2[B_N];
    int t = threadIdx.x;
    for (int b = t; b < B_N; b += 256)
        s2[b] = py2[(long long)b * L_N + l] / (c[l * B_N + b] + EPSF);
    __syncthreads();
    int wave = t >> 5, lane = t & 31;
    int a = blockIdx.y * 8 + wave;
    const float* ma = m + (long long)l * B_N * B_N + (long long)a * B_N;
    float s = 0.f;
    for (int b = lane; b < B_N; b += 32) s += ma[b] * s2[b];
    for (int off = 16; off > 0; off >>= 1) s += __shfl_xor(s, off, 32);
    if (lane == 0) {
        r[l * B_N + a] = s;
        if (fabsf(s - py1[(long long)a * L_N + l]) > ATOLF) atomicOr(&viol1[l], 1);
    }
}

// apply scaling (col, optionally row), write m, accumulate new colsums
__global__ __launch_bounds__(256)
void sink_scale_kernel(float* __restrict__ m, const float* __restrict__ c,
                       const float* __restrict__ r,
                       const float* __restrict__ py1, const float* __restrict__ py2,
                       float* __restrict__ c2, const int* __restrict__ viol1,
                       const int* __restrict__ done)
{
    int l = blockIdx.x;
    if (done[l]) return;
    __shared__ float s2[B_N];
    int t = threadIdx.x;
    for (int b = t; b < B_N; b += 256)
        s2[b] = py2[(long long)b * L_N + l] / (c[l * B_N + b] + EPSF);
    __syncthreads();
    int a0 = blockIdx.y * 64;
    float* ml = m + (long long)l * B_N * B_N;
    bool stop1 = (viol1[l] == 0);
    if (stop1) {
        // freeze on m1 = m * s2
        for (int a = a0; a < a0 + 64; ++a) {
            float* row = ml + (long long)a * B_N;
            for (int b = t; b < B_N; b += 256) row[b] *= s2[b];
        }
    } else {
        float colacc[8];
        for (int j = 0; j < 8; ++j) colacc[j] = 0.f;
        for (int a = a0; a < a0 + 64; ++a) {
            float s1v = py1[(long long)a * L_N + l] / (r[l * B_N + a] + EPSF);
            float* row = ml + (long long)a * B_N;
            for (int j = 0; j < 8; ++j) {
                int b = j * 256 + t;
                float v = row[b] * s2[b] * s1v;
                row[b] = v;
                colacc[j] += v;
            }
        }
        for (int j = 0; j < 8; ++j) atomicAdd(&c2[l * B_N + j * 256 + t], colacc[j]);
    }
}

__global__ void sink_finish_kernel(float* __restrict__ c, const float* __restrict__ c2,
                                   const float* __restrict__ py2,
                                   const int* __restrict__ viol1, int* __restrict__ done)
{
    int l = blockIdx.x, t = threadIdx.x;
    if (done[l]) return;
    int stop1 = (viol1[l] == 0);
    int viol = 0;
    for (int b = t; b < B_N; b += 256)
        if (fabsf(c2[l * B_N + b] - py2[(long long)b * L_N + l]) > ATOLF) viol = 1;
    __shared__ int red[256];
    red[t] = viol; __syncthreads();
    for (int s = 128; s > 0; s >>= 1) { if (t < s) red[t] |= red[t + s]; __syncthreads(); }
    int stop2 = (red[0] == 0);
    if (!stop1) for (int b = t; b < B_N; b += 256) c[l * B_N + b] = c2[l * B_N + b];
    if (t == 0 && (stop1 || stop2)) done[l] = 1;
}

// ---------------- final statistics ----------------
__global__ __launch_bounds__(256)
void rowsum_S_kernel(const float* __restrict__ m, float* __restrict__ S)
{
    int l = blockIdx.x;
    int t = threadIdx.x, wave = t >> 5, lane = t & 31;
    int a = blockIdx.y * 8 + wave;
    const float* ma = m + (long long)l * B_N * B_N + (long long)a * B_N;
    float s = 0.f;
    for (int b = lane; b < B_N; b += 32) s += ma[b];
    for (int off = 16; off > 0; off >>= 1) s += __shfl_xor(s, off, 32);
    if (lane == 0) S[l * B_N + a] = s;
}

__global__ void zero_accs_kernel(float* __restrict__ accs)
{
    if (blockIdx.x == 0 && threadIdx.x < 8) accs[threadIdx.x] = 0.f;
}

__global__ __launch_bounds__(256)
void final_loss_kernel(const float* __restrict__ m, const float* __restrict__ S,
                       const float* __restrict__ py1, const float* __restrict__ p_y,
                       float* __restrict__ alignOut, float* __restrict__ accs)
{
    int a = blockIdx.x, chunk = blockIdx.y, t = threadIdx.x;
    int b = chunk * 256 + t;
    __shared__ float py1s[L_N], lpy1s[L_N], invS[L_N], lpy[L_N];
    if (t < L_N) {
        float p = py1[(long long)a * L_N + t];
        py1s[t]  = p;
        lpy1s[t] = __logf(p + EPSF);
        invS[t]  = 1.f / (S[t * B_N + a] + EPSF);
        lpy[t]   = __logf(p_y[t] + EPSF);
    }
    __syncthreads();
    float q[L_N];
    float denom = 0.f;
    long long base = (long long)a * B_N + b;
    for (int l = 0; l < L_N; ++l) {
        float mv = m[(long long)l * B_N * B_N + base];
        alignOut[base * L_N + l] = mv;            // align output [a][b][l]
        q[l] = mv * invS[l];
        denom += q[l] * py1s[l];
    }
    float ldn = __logf(denom + EPSF);
    float aL = 0.f, aM = 0.f;
    for (int l = 0; l < L_N; ++l) {
        float lt = __logf(q[l] + EPSF) - ldn;
        float w  = py1s[l] * q[l];
        aL += w * lt;
        aM += w * (lt + lpy1s[l] - lpy[l]);
    }
    __shared__ float r1[256], r2[256];
    r1[t] = aL; r2[t] = aM; __syncthreads();
    for (int s = 128; s > 0; s >>= 1) {
        if (t < s) { r1[t] += r1[t + s]; r2[t] += r2[t + s]; }
        __syncthreads();
    }
    if (t == 0) { atomicAdd(&accs[0], r1[0]); atomicAdd(&accs[1], r2[0]); }
}

__global__ void mi_disc_kernel(const float* __restrict__ p1, const float* __restrict__ p2,
                               const float* __restrict__ p12, const float* __restrict__ p_y,
                               float* __restrict__ accs)
{
    int j = blockIdx.x, chunk = blockIdx.y, t = threadIdx.x;
    const float* p = (j == 0) ? p1 : (j == 1) ? p2 : p12;
    int a = chunk * 256 + t;
    float s = 0.f;
    for (int l = 0; l < L_N; ++l) {
        float v = p[(long long)a * L_N + l];
        s += v * (__logf(v) - __logf(p_y[l]));
    }
    __shared__ float red[256];
    red[t] = s; __syncthreads();
    for (int st = 128; st > 0; st >>= 1) { if (t < st) red[t] += red[t + st]; __syncthreads(); }
    if (t == 0) atomicAdd(&accs[2 + j], red[0]);
}

__global__ void write_out_kernel(const float* __restrict__ accs, float* __restrict__ out)
{
    if (blockIdx.x == 0 && threadIdx.x == 0) {
        float loss = accs[0] / (float)B_N;
        float miq  = accs[1] / (float)B_N;
        float mi1  = accs[2] / (float)B_N;
        float mi2  = accs[3] / (float)B_N;
        float mi12 = accs[4] / (float)B_N;
        out[0] = loss;
        out[1] = mi1 + mi2 - miq;   // redundancy
        out[2] = miq - mi2;         // unique1
        out[3] = miq - mi1;         // unique2
        out[4] = mi12 - miq;        // synergy
    }
}

// ---------------- host orchestration ----------------
extern "C" void kernel_launch(void* const* d_in, const int* in_sizes, int n_in,
                              void* d_out, int out_size, void* d_ws, size_t ws_size,
                              hipStream_t stream)
{
    (void)in_sizes; (void)n_in; (void)out_size; (void)ws_size;

    const float* x1  = (const float*)d_in[0];
    const float* x2  = (const float*)d_in[1];
    // d_in[2] = y (unused by reference outputs)
    const float* m1W[4] = { (const float*)d_in[3], (const float*)d_in[5],
                            (const float*)d_in[7], (const float*)d_in[9] };
    const float* m1b[4] = { (const float*)d_in[4], (const float*)d_in[6],
                            (const float*)d_in[8], (const float*)d_in[10] };
    const float* m2W[4] = { (const float*)d_in[11], (const float*)d_in[13],
                            (const float*)d_in[15], (const float*)d_in[17] };
    const float* m2b[4] = { (const float*)d_in[12], (const float*)d_in[14],
                            (const float*)d_in[16], (const float*)d_in[18] };
    const float* d1W  = (const float*)d_in[19];
    const float* d1b  = (const float*)d_in[20];
    const float* d2W  = (const float*)d_in[21];
    const float* d2b  = (const float*)d_in[22];
    const float* d12W = (const float*)d_in[23];
    const float* d12b = (const float*)d_in[24];
    const float* p_y  = (const float*)d_in[25];

    float* out = (float*)d_out;

    // workspace carve-out (256B aligned)
    char* ws = (char*)d_ws;
    auto alloc = [&](size_t bytes) -> void* {
        void* p = (void*)ws;
        ws += (bytes + 255) & ~(size_t)255;
        return p;
    };
    float*    mM   = (float*)alloc((size_t)L_N * B_N * B_N * 4);   // 168 MB sinkhorn state
    _Float16* x1h  = (_Float16*)alloc((size_t)B_N * D1_N * 2);
    _Float16* x2h  = (_Float16*)alloc((size_t)B_N * D2_N * 2);
    _Float16* hA   = (_Float16*)alloc((size_t)B_N * H_N * 2);
    _Float16* hB   = (_Float16*)alloc((size_t)B_N * H_N * 2);
    const int wsz[4] = { H_N * D1_N, H_N * H_N, H_N * H_N, Q_N * H_N };
    _Float16* W1h[4]; _Float16* W2h[4];
    for (int i = 0; i < 4; ++i) W1h[i] = (_Float16*)alloc((size_t)wsz[i] * 2);
    for (int i = 0; i < 4; ++i) W2h[i] = (_Float16*)alloc((size_t)wsz[i] * 2);
    float*    q1f  = (float*)alloc((size_t)B_N * Q_N * 4);
    float*    q2f  = (float*)alloc((size_t)B_N * Q_N * 4);
    _Float16* q1h  = (_Float16*)alloc((size_t)B_N * Q_N * 2);
    _Float16* q2h  = (_Float16*)alloc((size_t)B_N * Q_N * 2);
    float*    py1  = (float*)alloc((size_t)B_N * L_N * 4);
    float*    py2  = (float*)alloc((size_t)B_N * L_N * 4);
    float*    py12 = (float*)alloc((size_t)B_N * L_N * 4);
    float*    cS   = (float*)alloc((size_t)L_N * B_N * 4);
    float*    c2S  = (float*)alloc((size_t)L_N * B_N * 4);
    float*    rS   = (float*)alloc((size_t)L_N * B_N * 4);
    float*    Ssum = (float*)alloc((size_t)L_N * B_N * 4);
    int*      viol1 = (int*)alloc((size_t)L_N * 4);
    int*      done  = (int*)alloc((size_t)L_N * 4);
    float*    accs  = (float*)alloc(8 * 4);

    dim3 blk256(256);

    auto conv = [&](const float* src, _Float16* dst, int n) {
        f32_to_f16_kernel<<<dim3((n + 255) / 256), blk256, 0, stream>>>(src, dst, n);
    };
    auto gemm = [&](const _Float16* A, int lda, long long sA,
                    const _Float16* Bp, int ldb, long long sB,
                    const float* bias, float* oF, _Float16* oH, int ldc, long long sC,
                    int M, int N, int K, int mode, float scale, int batch) {
        dim3 grid(N / BN, M / BM, batch);
        wmma_gemm_kernel<<<grid, blk256, 0, stream>>>(A, lda, sA, Bp, ldb, sB, bias,
                                                      oF, oH, ldc, sC, K, mode, scale);
    };

    // 1) precision conversion for the WMMA path
    conv(x1, x1h, B_N * D1_N);
    conv(x2, x2h, B_N * D2_N);
    for (int i = 0; i < 4; ++i) conv(m1W[i], W1h[i], wsz[i]);
    for (int i = 0; i < 4; ++i) conv(m2W[i], W2h[i], wsz[i]);

    // 2) discriminators (+softmax), f32 VALU (tiny: L=10)
    disc_softmax_kernel<<<dim3(B_N), blk256, 0, stream>>>(x1, D1_N, x1, 0, d1W, d1b, py1);
    disc_softmax_kernel<<<dim3(B_N), blk256, 0, stream>>>(x2, D2_N, x2, 0, d2W, d2b, py2);
    disc_softmax_kernel<<<dim3(B_N), blk256, 0, stream>>>(x1, D1_N, x2, D2_N, d12W, d12b, py12);

    // 3) MLP1: 512 -> 1024 -> 1024 -> 1024 -> 640 (WMMA f16, f32 accum)
    gemm(x1h, D1_N, 0, W1h[0], D1_N, 0, m1b[0], nullptr, hA, H_N, 0, B_N, H_N, D1_N, 1, 0.f, 1);
    gemm(hA,  H_N, 0, W1h[1], H_N, 0, m1b[1], nullptr, hB, H_N, 0, B_N, H_N, H_N, 1, 0.f, 1);
    gemm(hB,  H_N, 0, W1h[2], H_N, 0, m1b[2], nullptr, hA, H_N, 0, B_N, H_N, H_N, 1, 0.f, 1);
    gemm(hA,  H_N, 0, W1h[3], H_N, 0, m1b[3], q1f, nullptr, Q_N, 0, B_N, Q_N, H_N, 0, 0.f, 1);
    // MLP2
    gemm(x2h, D2_N, 0, W2h[0], D2_N, 0, m2b[0], nullptr, hA, H_N, 0, B_N, H_N, D2_N, 1, 0.f, 1);
    gemm(hA,  H_N, 0, W2h[1], H_N, 0, m2b[1], nullptr, hB, H_N, 0, B_N, H_N, H_N, 1, 0.f, 1);
    gemm(hB,  H_N, 0, W2h[2], H_N, 0, m2b[2], nullptr, hA, H_N, 0, B_N, H_N, H_N, 1, 0.f, 1);
    gemm(hA,  H_N, 0, W2h[3], H_N, 0, m2b[3], q2f, nullptr, Q_N, 0, B_N, Q_N, H_N, 0, 0.f, 1);

    // 4) per-(a,l) normalization over E=64 (ddof=1) -> f16 for einsum
    norm_rows_kernel<<<dim3(B_N * L_N), dim3(64), 0, stream>>>(q1f, q1h);
    norm_rows_kernel<<<dim3(B_N * L_N), dim3(64), 0, stream>>>(q2f, q2h);

    // 5) batched einsum per label: m[l] = exp( (q1_l @ q2_l^T) / sqrt(64) )
    gemm(q1h, Q_N, E_N, q2h, Q_N, E_N, nullptr, mM, nullptr,
         B_N, (long long)B_N * B_N, B_N, B_N, E_N, 2, 0.125f, L_N);

    // 6) Sinkhorn (L2-resident): faithful frozen-carry early stop per label
    sink_init_colsum_kernel<<<dim3(L_N, B_N / 256), blk256, 0, stream>>>(mM, cS, done);
    for (int it = 0; it < MAX_ITERS; ++it) {
        sink_reset_kernel<<<dim3(L_N), blk256, 0, stream>>>(c2S, viol1);
        sink_rowsum_kernel<<<dim3(L_N, B_N / 8), blk256, 0, stream>>>(mM, cS, py1, py2, rS, viol1, done);
        sink_scale_kernel<<<dim3(L_N, B_N / 64), blk256, 0, stream>>>(mM, cS, rS, py1, py2, c2S, viol1, done);
        sink_finish_kernel<<<dim3(L_N), blk256, 0, stream>>>(cS, c2S, py2, viol1, done);
    }

    // 7) final statistics + align output
    rowsum_S_kernel<<<dim3(L_N, B_N / 8), blk256, 0, stream>>>(mM, Ssum);
    zero_accs_kernel<<<dim3(1), dim3(32), 0, stream>>>(accs);
    final_loss_kernel<<<dim3(B_N, B_N / 256), blk256, 0, stream>>>(mM, Ssum, py1, p_y, out + 5, accs);
    mi_disc_kernel<<<dim3(3, B_N / 256), blk256, 0, stream>>>(py1, py2, py12, p_y, accs);
    write_out_kernel<<<dim3(1), dim3(32), 0, stream>>>(accs, out);
}
